// GCN_EdgeClassifier_38027640439084
// MI455X (gfx1250) — compile-verified, matrix operand order checked by reading, not run on hardware
//
#include <hip/hip_runtime.h>
#include <hip/hip_bf16.h>

#define NN 10000     // nodes
#define NE 160000    // edges
#define IND 128
#define HID 512

typedef __attribute__((ext_vector_type(16))) __bf16 bf16x16;
typedef __attribute__((ext_vector_type(8)))  __bf16 bf16x8;
typedef __attribute__((ext_vector_type(8)))  float  f32x8;
typedef __attribute__((ext_vector_type(4)))  unsigned int u32x4;
typedef __attribute__((ext_vector_type(8)))  unsigned int u32x8;

__device__ __forceinline__ unsigned short f2bf(float f) {
  unsigned u = __float_as_uint(f);
  u += 0x7FFFu + ((u >> 16) & 1u);           // round-to-nearest-even
  return (unsigned short)(u >> 16);
}

__device__ __forceinline__ bf16x16 cat8(bf16x8 lo, bf16x8 hi) {
  return __builtin_shufflevector(lo, hi, 0,1,2,3,4,5,6,7,8,9,10,11,12,13,14,15);
}

// Global half-fragment pair: 8 bf16 at p, 8 bf16 at p+16
__device__ __forceinline__ bf16x16 load_frag_g(const unsigned short* p) {
  bf16x8 lo = *(const bf16x8*)p;
  bf16x8 hi = *(const bf16x8*)(p + 16);
  return cat8(lo, hi);
}

// TDM LDS padding: +16B after every 1KB of payload
__device__ __forceinline__ unsigned lds_pad(unsigned u) {
  return u + ((u >> 10) << 4);
}

// ---------------------------------------------------------------------------
// Tiled bf16 WMMA GEMM: C[M,N] = A[M,K] @ BT[N,K]^T (+ bias)
// Block = 256 thr (8 waves), tile 256(M) x 64(N); each wave owns two 16-row
// subtiles (acc[2][4]) so each LDS B-fragment feeds 2 WMMAs.
// The 64xK weight tile is DMA'd to LDS once per block by the Tensor Data
// Mover (tensor_load_to_lds, D# per ISA 8.3/8.4, HW padding for bank spread).
// Padded LDS addresses are decomposed into loop-invariant row bases + one
// shared per-k term, so the inner loop is WMMA/ds-bound, not VALU-bound.
// GATHER=1: A row e is [hb[src[e]], hb[dst[e]]].
// ---------------------------------------------------------------------------
template<int GATHER>
__global__ void gemm_bf16_kernel(const unsigned short* __restrict__ A,
                                 const unsigned short* __restrict__ HB,
                                 const int* __restrict__ esrc,
                                 const int* __restrict__ edst,
                                 const unsigned short* __restrict__ BT,
                                 const float* __restrict__ bias,
                                 float* __restrict__ C,
                                 int M, int N, int K) {
  extern __shared__ char smem[];               // 64*K*2 + pad bytes
  const int lane = threadIdx.x & 31;
  const int wave = threadIdx.x >> 5;
  const int half = lane >> 4;     // K-chunk selector for A/B fragments
  const int lr   = lane & 15;     // row (A) / col (B,C) within 16-tile
  const int n0   = blockIdx.x * 64;
  const int mbase = blockIdx.y * 256 + wave * 16;   // subtile t adds t*128

  // ---- TDM: one descriptor loads BT[n0 .. n0+63][0..K) into LDS ----------
  if (threadIdx.x < 32) {
    unsigned lds_off =
        (unsigned)(size_t)(__attribute__((address_space(3))) char*)smem;
    unsigned long long ga =
        (unsigned long long)(size_t)(BT + (size_t)n0 * K);
    u32x4 g0;
    g0.x = 1u;                                  // count=1 (valid descriptor)
    g0.y = lds_off;                             // lds_addr
    g0.z = (unsigned)ga;                        // global_addr[31:0]
    g0.w = (unsigned)((ga >> 32) & 0x1FFFFFFu)  // global_addr[56:32]
           | 0x80000000u;                       // type=2 ("image")
    u32x8 g1;
    g1[0] = (1u << 16)        // data_size = 2 bytes
          | (1u << 20)        // pad_enable
          | (7u << 22)        // pad_interval: 1024B
          | (3u << 25);       // pad_amount: 4 dwords = 16B
    g1[1] = ((unsigned)K & 0xFFFFu) << 16;            // tensor_dim0[15:0]
    g1[2] = (64u << 16) | ((unsigned)K >> 16);        // dim1[15:0] | dim0 hi
    g1[3] = ((unsigned)K & 0xFFFFu) << 16;            // tile_dim0 = K
    g1[4] = 64u;                                      // tile_dim1 = 64 rows
    g1[5] = (unsigned)K;                              // tensor_dim0_stride
    g1[6] = 0u;
    g1[7] = 0u;
    asm volatile("tensor_load_to_lds %0, %1" : : "s"(g0), "s"(g1) : "memory");
    __builtin_amdgcn_s_wait_tensorcnt(0);
  }
  __syncthreads();

  // Loop-invariant LDS bases per N-subtile (row stride = 2K + pads)
  unsigned bbase[4];
#pragma unroll
  for (int j = 0; j < 4; ++j) {
    unsigned u = (unsigned)(j * 16 + lr) * (unsigned)(K * 2);
    bbase[j] = lds_pad(u) + (unsigned)(half * 16);
  }

  // Clamped A rows (branch-free OOB handling; stores are guarded)
  int mr[2];
  const unsigned short* arow[2];
  int nS[2], nD[2];
#pragma unroll
  for (int t = 0; t < 2; ++t) {
    int m = mbase + t * 128 + lr;
    mr[t] = (m < M) ? m : (M - 1);
    if (GATHER) { nS[t] = esrc[mr[t]]; nD[t] = edst[mr[t]]; }
    else        { arow[t] = A + (size_t)mr[t] * K + half * 8; }
  }

  f32x8 acc[2][4];
#pragma unroll
  for (int j = 0; j < 4; ++j) {
    float bv = bias ? bias[n0 + j * 16 + lr] : 0.0f;
#pragma unroll
    for (int t = 0; t < 2; ++t)
#pragma unroll
      for (int r = 0; r < 8; ++r) acc[t][j][r] = bv;
  }

  for (int k0 = 0; k0 < K; k0 += 32) {
    unsigned kb = (unsigned)k0 * 2u;
    unsigned pk = kb + ((kb >> 10) << 4);   // shared per-k padded offset
    bf16x16 a[2];
#pragma unroll
    for (int t = 0; t < 2; ++t) {
      const unsigned short* ap;
      if (GATHER) {
        int node = (k0 < HID) ? nS[t] : nD[t];  // 32-chunks never straddle 512
        ap = HB + (size_t)node * HID + (k0 & (HID - 1)) + half * 8;
      } else {
        ap = arow[t] + k0;
      }
      a[t] = load_frag_g(ap);
    }
#pragma unroll
    for (int j = 0; j < 4; ++j) {
      const char* bp = smem + (bbase[j] + pk);
      bf16x8 blo = *(const bf16x8*)bp;
      bf16x8 bhi = *(const bf16x8*)(bp + 32);
      bf16x16 b = cat8(blo, bhi);
#pragma unroll
      for (int t = 0; t < 2; ++t)
        acc[t][j] = __builtin_amdgcn_wmma_f32_16x16x32_bf16(
            false, a[t], false, b, (short)0, acc[t][j], false, false);
    }
  }

#pragma unroll
  for (int t = 0; t < 2; ++t)
#pragma unroll
    for (int j = 0; j < 4; ++j)
#pragma unroll
      for (int r = 0; r < 8; ++r) {
        int m = mbase + t * 128 + r + 8 * half;
        if (m < M) C[(size_t)m * N + n0 + j * 16 + lr] = acc[t][j][r];
      }
}

// ---------------------------------------------------------------------------
// Degree / normalization
// ---------------------------------------------------------------------------
__global__ void deg_init_kernel(float* deg, int n) {
  int i = blockIdx.x * blockDim.x + threadIdx.x;
  if (i < n) deg[i] = 1.0f;                       // self-loop
}
__global__ void deg_count_kernel(float* deg, const int* ei, int e) {
  int i = blockIdx.x * blockDim.x + threadIdx.x;
  if (i < e) atomicAdd(&deg[ei[e + i]], 1.0f);    // col = targets
}
__global__ void deg_rsqrt_kernel(float* deg, int n) {
  int i = blockIdx.x * blockDim.x + threadIdx.x;
  if (i < n) deg[i] = rsqrtf(deg[i]);
}

// agg[n,c] = dis[n]^2 * tmp[n,c] + bias[c]   (self-loop + bias init)
__global__ void self_bias_kernel(const float* __restrict__ dis,
                                 const float* __restrict__ tmp,
                                 const float* __restrict__ bias,
                                 float* __restrict__ agg) {
  int idx = blockIdx.x * 256 + threadIdx.x;       // over NN*HID
  int n = idx >> 9, c = idx & (HID - 1);
  float w = dis[n];
  agg[idx] = w * w * tmp[idx] + bias[c];
}

// agg[dst] += dis[src]*dis[dst]*tmp[src]; one block per edge, 2 ch/thread
__global__ void scatter_kernel(const int* __restrict__ ei,
                               const float* __restrict__ dis,
                               const float* __restrict__ tmp,
                               float* __restrict__ agg, int E) {
  int e = blockIdx.x, c = threadIdx.x;
  int s = ei[e], d = ei[E + e];
  float w = dis[s] * dis[d];
  const float* tr = tmp + (size_t)s * HID;
  float* ar = agg + (size_t)d * HID;
  atomicAdd(&ar[c],       w * tr[c]);
  atomicAdd(&ar[c + 256], w * tr[c + 256]);
}

// ---------------------------------------------------------------------------
// LayerNorm (+ReLU before or after); block = 256 = one row; W in {512,1024}
// ---------------------------------------------------------------------------
template<bool RELU_FIRST, bool OUT_BF16>
__global__ void ln_kernel(const float* __restrict__ in,
                          const float* __restrict__ g,
                          const float* __restrict__ b,
                          float* __restrict__ outf,
                          unsigned short* __restrict__ outb, int W) {
  __shared__ float ssum[256], ssq[256];
  const size_t row = blockIdx.x;
  const int tid = threadIdx.x;
  float vals[4];
  float s = 0.f, q = 0.f;
#pragma unroll
  for (int i = 0; i < 4; ++i) {
    int c = tid + i * 256;
    if (c < W) {
      float v = in[row * W + c];
      if (RELU_FIRST) v = fmaxf(v, 0.f);
      vals[i] = v; s += v; q += v * v;
    }
  }
  ssum[tid] = s; ssq[tid] = q;
  __syncthreads();
  for (int o = 128; o > 0; o >>= 1) {
    if (tid < o) { ssum[tid] += ssum[tid + o]; ssq[tid] += ssq[tid + o]; }
    __syncthreads();
  }
  float mu = ssum[0] / W;
  float var = ssq[0] / W - mu * mu;
  float rs = rsqrtf(var + 1e-5f);
#pragma unroll
  for (int i = 0; i < 4; ++i) {
    int c = tid + i * 256;
    if (c < W) {
      float y = (vals[i] - mu) * rs * g[c] + b[c];
      if (!RELU_FIRST) y = fmaxf(y, 0.f);
      if (OUT_BF16) outb[row * W + c] = f2bf(y);
      else          outf[row * W + c] = y;
    }
  }
}

// ---------------------------------------------------------------------------
// Weight prep and activation conversion
// ---------------------------------------------------------------------------
__global__ void transpose_bf16_kernel(const float* __restrict__ W,
                                      unsigned short* __restrict__ WT,
                                      int K, int N) {
  int idx = blockIdx.x * 256 + threadIdx.x;       // over K*N
  if (idx < K * N) {
    int k = idx / N, n = idx - k * N;
    WT[(size_t)n * K + k] = f2bf(W[idx]);
  }
}
__global__ void f32_to_bf16_kernel(const float* __restrict__ in,
                                   unsigned short* __restrict__ out, int n) {
  int i = blockIdx.x * 256 + threadIdx.x;
  if (i < n) out[i] = f2bf(in[i]);
}

// ---------------------------------------------------------------------------
// Head: out[e] = dot(z2[e,:512], mw2) + mb2 ; one wave32 per edge
// ---------------------------------------------------------------------------
__global__ void head_kernel(const float* __restrict__ z,
                            const float* __restrict__ w,
                            const float* __restrict__ b2,
                            float* __restrict__ out, int E) {
  int wv = threadIdx.x >> 5, lane = threadIdx.x & 31;
  int e = blockIdx.x * 8 + wv;
  if (e >= E) return;
  const float* zr = z + (size_t)e * HID;
  float s = 0.f;
#pragma unroll
  for (int i = 0; i < HID / 32; ++i) s += zr[lane + i * 32] * w[lane + i * 32];
#pragma unroll
  for (int o = 16; o > 0; o >>= 1) s += __shfl_xor(s, o, 32);
  if (lane == 0) out[e] = s + b2[0];
}

// ---------------------------------------------------------------------------
extern "C" void kernel_launch(void* const* d_in, const int* in_sizes, int n_in,
                              void* d_out, int out_size, void* d_ws, size_t ws_size,
                              hipStream_t stream) {
  const float* x    = (const float*)d_in[0];
  const int*   ei   = (const int*)  d_in[1];
  const float* w0   = (const float*)d_in[2];
  const float* b0   = (const float*)d_in[3];
  const float* wsA  = (const float*)d_in[4];   // [3,512,512]
  const float* bsA  = (const float*)d_in[5];   // [3,512]
  const float* lng  = (const float*)d_in[6];   // [4,512]
  const float* lnb  = (const float*)d_in[7];
  const float* mw0  = (const float*)d_in[8];   // [1024,1024]
  const float* mb0  = (const float*)d_in[9];
  const float* mg0  = (const float*)d_in[10];
  const float* mt0  = (const float*)d_in[11];
  const float* mw1  = (const float*)d_in[12];  // [1024,512]
  const float* mb1  = (const float*)d_in[13];
  const float* mg1  = (const float*)d_in[14];
  const float* mt1  = (const float*)d_in[15];
  const float* mw2  = (const float*)d_in[16];  // [512,1]
  const float* mb2  = (const float*)d_in[17];
  float* out = (float*)d_out;

  size_t off = 0;
  char* base = (char*)d_ws;
  auto carve = [&](size_t bytes) -> char* {
    char* p = base + off;
    off = (off + bytes + 255) & ~(size_t)255;
    return p;
  };
  float*          dis   = (float*)         carve((size_t)NN * 4);
  unsigned short* wt0   = (unsigned short*)carve((size_t)IND * HID * 2);
  unsigned short* wt1   = (unsigned short*)carve((size_t)HID * HID * 2);
  unsigned short* wt2   = (unsigned short*)carve((size_t)HID * HID * 2);
  unsigned short* wt3   = (unsigned short*)carve((size_t)HID * HID * 2);
  unsigned short* mw0T  = (unsigned short*)carve((size_t)1024 * 1024 * 2);
  unsigned short* mw1T  = (unsigned short*)carve((size_t)512 * 1024 * 2);
  unsigned short* hb    = (unsigned short*)carve((size_t)NN * HID * 2);
  float*          tmp   = (float*)         carve((size_t)NN * HID * 4);
  float*          agg   = (float*)         carve((size_t)NN * HID * 4);
  float*          z1    = (float*)         carve((size_t)NE * 1024 * 4); // z2 aliases
  unsigned short* z1b   = (unsigned short*)carve((size_t)NE * 1024 * 2);
  float*          z2    = z1;                       // z1 dead once z1b written

  // LDS bytes for the TDM-staged 64xK weight tile (+16B pad per 1KB)
  auto smem_sz = [](int K) -> size_t {
    unsigned bytes = 64u * (unsigned)K * 2u;
    return (size_t)(bytes + ((bytes >> 10) << 4));
  };

  // 1) symmetric normalization coefficients
  deg_init_kernel <<<(NN + 255) / 256, 256, 0, stream>>>(dis, NN);
  deg_count_kernel<<<(NE + 255) / 256, 256, 0, stream>>>(dis, ei, NE);
  deg_rsqrt_kernel<<<(NN + 255) / 256, 256, 0, stream>>>(dis, NN);

  // 2) weights -> bf16, transposed to [N][K]
  transpose_bf16_kernel<<<(IND * HID) / 256, 256, 0, stream>>>(w0, wt0, IND, HID);
  unsigned short* wts[3] = {wt1, wt2, wt3};
  for (int i = 0; i < 3; ++i)
    transpose_bf16_kernel<<<(HID * HID) / 256, 256, 0, stream>>>(
        wsA + (size_t)i * HID * HID, wts[i], HID, HID);
  transpose_bf16_kernel<<<(1024 * 1024) / 256, 256, 0, stream>>>(mw0, mw0T, 1024, 1024);
  transpose_bf16_kernel<<<(1024 * 512) / 256, 256, 0, stream>>>(mw1, mw1T, 1024, 512);

  // 3) layer-0 input
  f32_to_bf16_kernel<<<(NN * IND + 255) / 256, 256, 0, stream>>>(x, hb, NN * IND);

  // 4) GCN layers
  const int mblocks = (NN + 255) / 256;   // 40
  for (int i = 0; i < 4; ++i) {
    int K = (i == 0) ? IND : HID;
    const unsigned short* WT = (i == 0) ? wt0 : wts[i - 1];
    const float* bias = (i == 0) ? b0 : bsA + (size_t)(i - 1) * HID;
    gemm_bf16_kernel<0><<<dim3(HID / 64, mblocks), 256, smem_sz(K), stream>>>(
        hb, nullptr, nullptr, nullptr, WT, nullptr, tmp, NN, HID, K);
    self_bias_kernel<<<(NN * HID) / 256, 256, 0, stream>>>(dis, tmp, bias, agg);
    scatter_kernel  <<<NE, 256, 0, stream>>>(ei, dis, tmp, agg, NE);
    ln_kernel<true, true><<<NN, 256, 0, stream>>>(
        agg, lng + (size_t)i * HID, lnb + (size_t)i * HID, nullptr, hb, HID);
  }

  // 5) edge classifier: GEMM1 with fused [h[src],h[dst]] gather
  gemm_bf16_kernel<1><<<dim3(1024 / 64, NE / 256), 256, smem_sz(1024), stream>>>(
      nullptr, hb, ei, ei + NE, mw0T, mb0, z1, NE, 1024, 1024);
  ln_kernel<false, true><<<NE, 256, 0, stream>>>(z1, mg0, mt0, nullptr, z1b, 1024);

  gemm_bf16_kernel<0><<<dim3(512 / 64, NE / 256), 256, smem_sz(1024), stream>>>(
      z1b, nullptr, nullptr, nullptr, mw1T, mb1, z2, NE, 512, 1024);
  ln_kernel<false, false><<<NE, 256, 0, stream>>>(z2, mg1, mt1, z2, nullptr, 512);

  // 6) head
  head_kernel<<<NE / 8, 256, 0, stream>>>(z2, mw2, mb2, out, NE);
}
